// DualAttentionModule_38989713113557
// MI455X (gfx1250) — compile-verified, matrix operand order checked by reading, not run on hardware
//
#include <hip/hip_runtime.h>

// ---------------------------------------------------------------------------
// DANet dual attention (PAM + CAM) for gfx1250, bf16 WMMA with f32 accumulate.
// B=4, H=W=64, C=512, N=H*W=4096, K8=C/8=64.
// All hot WMMA operands use adjacent-K-pair layouts so loads coalesce to b128.
// ---------------------------------------------------------------------------

#define BATCH 4
#define NTOK  4096        // H*W
#define CCH   512
#define K8    64
#define PLANE (NTOK * CCH)      // 2097152 elements per batch

typedef __attribute__((ext_vector_type(16))) __bf16 v16bf;
typedef __attribute__((ext_vector_type(8)))  float  v8f;
typedef __attribute__((ext_vector_type(4)))  float  v4f;
typedef __attribute__((ext_vector_type(4)))  unsigned int v4u;

union BF16x16 {
    v16bf v;
    unsigned int  u[8];
    unsigned short h[16];
};

__device__ __forceinline__ unsigned short f2bf(float f) {
    unsigned int u = __float_as_uint(f);
    u += 0x7FFFu + ((u >> 16) & 1u);   // round-to-nearest-even
    return (unsigned short)(u >> 16);
}

// K offset pattern for 16-bit A/B operands (ISA 7.12.2): lanes 0-15 use K base 0,
// lanes 16-31 use K base 8; VGPRs 0..3 cover K[kb..kb+7], VGPRs 4..7 cover K[kb+16..kb+23].
__device__ __forceinline__ int wmma_k_off(int v, int half) {
    int kb = half << 3;
    return (v < 4) ? (kb + 2 * v) : (kb + 16 + 2 * (v - 4));
}

// Tile loader for row-major [R,K] bf16: lane owns row m, adjacent K pairs -> b32
// loads (compiler merges into b64/b128). Used for both A and B operands (B is
// fed from a transposed copy so its K runs contiguously too).
__device__ __forceinline__ v16bf load_tile_pairs(const unsigned short* base, int ldk,
                                                 int m, int k0, int half) {
    BF16x16 r;
    const unsigned short* row = base + (size_t)m * ldk + k0;
#pragma unroll
    for (int v = 0; v < 8; ++v) {
        int kk = wmma_k_off(v, half);
        r.u[v] = *(const unsigned int*)(row + kk);
    }
    return r.v;
}

__device__ __forceinline__ v8f wmma_bf16(v16bf a, v16bf b, v8f c) {
    return __builtin_amdgcn_wmma_f32_16x16x32_bf16(false, a, false, b, (short)0, c,
                                                   false, false);
}

// --------------------- converts (x also emits transpose) -------------------
__global__ void k_convert_x(const float* __restrict__ src,
                            unsigned short* __restrict__ xb,
                            unsigned short* __restrict__ xT) {
    for (int i = blockIdx.x * blockDim.x + threadIdx.x; i < BATCH * PLANE;
         i += gridDim.x * blockDim.x) {
        unsigned short h = f2bf(src[i]);
        xb[i] = h;
        int b = i / PLANE, rem = i % PLANE;
        int n = rem / CCH,  c = rem % CCH;
        xT[(size_t)b * PLANE + (size_t)c * NTOK + n] = h;
    }
}

__global__ void k_convert_wT(const float* __restrict__ w,
                             unsigned short* __restrict__ wT, int K, int Nc) {
    for (int i = blockIdx.x * blockDim.x + threadIdx.x; i < K * Nc;
         i += gridDim.x * blockDim.x) {
        int k = i / Nc, n = i % Nc;
        wT[(size_t)n * K + k] = f2bf(w[i]);
    }
}

// ------------------------- generic projection GEMM -------------------------
// Out = A[M,K] @ W[K,Nc], with W supplied transposed (WT[Nc,K]).
// outT=0: Out row-major [M,Nc].  outT=1: Out transposed [Nc,M] (OutLd = M),
// written as one packed b128 per lane.
__global__ void k_gemm_bf16(const unsigned short* __restrict__ A,
                            const unsigned short* __restrict__ WT,
                            unsigned short* __restrict__ Out,
                            int Mtiles, int Ntiles, int K, int OutLd, int outT) {
    int wave = (blockIdx.x * blockDim.x + threadIdx.x) >> 5;
    int lane = threadIdx.x & 31;
    int half = lane >> 4;
    int ln16 = lane & 15;
    if (wave >= Mtiles * Ntiles) return;
    int m0 = (wave / Ntiles) << 4;
    int n0 = (wave % Ntiles) << 4;

    v8f acc = {};
    for (int k0 = 0; k0 < K; k0 += 32) {
        v16bf a = load_tile_pairs(A  + (size_t)m0 * K, K, ln16, k0, half);
        v16bf b = load_tile_pairs(WT + (size_t)n0 * K, K, ln16, k0, half);
        acc = wmma_bf16(a, b, acc);
    }
    if (outT) {
        v4u p;
#pragma unroll
        for (int rr = 0; rr < 4; ++rr)
            p[rr] = (unsigned)f2bf(acc[2 * rr]) | ((unsigned)f2bf(acc[2 * rr + 1]) << 16);
        *(v4u*)(Out + (size_t)(n0 + ln16) * OutLd + m0 + (half << 3)) = p;
    } else {
#pragma unroll
        for (int r = 0; r < 8; ++r) {
            int m = m0 + r + (half << 3);
            Out[(size_t)m * OutLd + n0 + ln16] = f2bf(acc[r]);
        }
    }
}

// ------------------------- PAM pass 1: row stats ---------------------------
// Per-lane online max/sum-exp over that lane's share of the 4096 keys; one
// exact (M,L) merge across the 16-lane half at the end.
__global__ void k_pam_stats(const unsigned short* __restrict__ q,
                            const unsigned short* __restrict__ km,
                            float* __restrict__ stats) {
    int wave = (blockIdx.x * blockDim.x + threadIdx.x) >> 5;
    int lane = threadIdx.x & 31;
    int half = lane >> 4, ln16 = lane & 15;
    if (wave >= BATCH * (NTOK / 16)) return;
    int b  = wave >> 8;
    int i0 = (wave & 255) << 4;

    const unsigned short* qb = q + (size_t)(b * NTOK + i0) * K8;
    v16bf a0 = load_tile_pairs(qb, K8, ln16, 0, half);
    v16bf a1 = load_tile_pairs(qb, K8, ln16, 32, half);

    float runM[8], runL[8];
#pragma unroll
    for (int r = 0; r < 8; ++r) { runM[r] = -3.0e38f; runL[r] = 0.0f; }

    for (int jt = 0; jt < NTOK / 16; ++jt) {
        const unsigned short* kb = km + (size_t)(b * NTOK + (jt << 4)) * K8;
        v16bf b0 = load_tile_pairs(kb, K8, ln16, 0, half);
        v16bf b1 = load_tile_pairs(kb, K8, ln16, 32, half);
        v8f s = {};
        s = wmma_bf16(a0, b0, s);
        s = wmma_bf16(a1, b1, s);
#pragma unroll
        for (int r = 0; r < 8; ++r) {            // per-lane online update
            float t  = s[r];
            float nm = fmaxf(runM[r], t);
            runL[r]  = runL[r] * __expf(runM[r] - nm) + __expf(t - nm);
            runM[r]  = nm;
        }
    }
#pragma unroll
    for (int r = 0; r < 8; ++r) {                // half-wave (M,L) merge
        float M = runM[r], L = runL[r];
#pragma unroll
        for (int off = 1; off < 16; off <<= 1) {
            float Mo = __shfl_xor(M, off);
            float Lo = __shfl_xor(L, off);
            float nm = fmaxf(M, Mo);
            L = L * __expf(M - nm) + Lo * __expf(Mo - nm);
            M = nm;
        }
        if (ln16 == 0) {
            int i = i0 + r + (half << 3);
            stats[(size_t)(b * NTOK + i) * 2 + 0] = M;
            stats[(size_t)(b * NTOK + i) * 2 + 1] = L;
        }
    }
}

// ------------------------- PAM pass 2: attn @ d ----------------------------
// One block (8 waves) per 16-row query block. Each wave computes one 16-wide
// probability tile -> LDS (layout transpose) -> all waves WMMA against dT for
// their 64 output channels. dT is [C,N] so B-tiles are contiguous pair loads.
__global__ void k_pam_apply(const unsigned short* __restrict__ q,
                            const unsigned short* __restrict__ kmat,
                            const unsigned short* __restrict__ dT,
                            const float* __restrict__ stats,
                            const float* __restrict__ gamma_pam,
                            float* __restrict__ out) {
    __shared__ __align__(16) unsigned short sP[16 * 128];

    int b  = blockIdx.x >> 8;
    int i0 = (blockIdx.x & 255) << 4;
    int w    = threadIdx.x >> 5;
    int lane = threadIdx.x & 31;
    int half = lane >> 4, ln16 = lane & 15;

    const unsigned short* qb = q + (size_t)(b * NTOK + i0) * K8;
    v16bf a0 = load_tile_pairs(qb, K8, ln16, 0, half);
    v16bf a1 = load_tile_pairs(qb, K8, ln16, 32, half);

    float mrow[8], lrow[8];
#pragma unroll
    for (int r = 0; r < 8; ++r) {
        int i = i0 + r + (half << 3);
        mrow[r] = stats[(size_t)(b * NTOK + i) * 2 + 0];
        lrow[r] = stats[(size_t)(b * NTOK + i) * 2 + 1];
    }

    v8f acc[4];
#pragma unroll
    for (int ct = 0; ct < 4; ++ct) acc[ct] = (v8f){};

    const unsigned short* dbase = dT + (size_t)b * PLANE;   // [C][N]

    for (int js = 0; js < NTOK / 128; ++js) {
        int j0 = js * 128;
        int jw = j0 + (w << 4);
        const unsigned short* kb = kmat + (size_t)(b * NTOK + jw) * K8;
        // prefetch next j-step's key rows while this step computes
        __builtin_prefetch(kb + 128 * K8, 0, 1);
        v16bf b0 = load_tile_pairs(kb, K8, ln16, 0, half);
        v16bf b1 = load_tile_pairs(kb, K8, ln16, 32, half);
        v8f s = {};
        s = wmma_bf16(a0, b0, s);
        s = wmma_bf16(a1, b1, s);
#pragma unroll
        for (int r = 0; r < 8; ++r) {
            int row = r + (half << 3);
            sP[row * 128 + (w << 4) + ln16] = f2bf(__expf(s[r] - mrow[r]));
        }
        __syncthreads();
#pragma unroll
        for (int kc = 0; kc < 4; ++kc) {
            BF16x16 a2;
#pragma unroll
            for (int v = 0; v < 8; ++v) {
                int kk = wmma_k_off(v, half);
                a2.u[v] = *(const unsigned int*)(&sP[ln16 * 128 + (kc << 5) + kk]);
            }
#pragma unroll
            for (int ct = 0; ct < 4; ++ct) {
                int c0 = (w << 6) + (ct << 4);
                // B[k=j_local, n=c] = dT[c][j0 + kc*32 + k] : contiguous pairs
                v16bf bt = load_tile_pairs(dbase + (size_t)c0 * NTOK + j0 + (kc << 5),
                                           NTOK, ln16, 0, half);
                acc[ct] = wmma_bf16(a2.v, bt, acc[ct]);
            }
        }
        __syncthreads();
    }

    float g = gamma_pam[0];
#pragma unroll
    for (int ct = 0; ct < 4; ++ct) {
        int c = (w << 6) + (ct << 4) + ln16;
        float* ob = out + (size_t)b * PLANE + (size_t)c * NTOK + i0 + (half << 3);
        v4f lo, hi;
#pragma unroll
        for (int r = 0; r < 4; ++r) {
            lo[r] = g * acc[ct][r]     / lrow[r];
            hi[r] = g * acc[ct][r + 4] / lrow[r + 4];
        }
        // feat_p reshape: flat = c*N + i (reference's direct (C,N)->HWC reshape)
        *(v4f*)(ob)     = lo;
        *(v4f*)(ob + 4) = hi;
    }
}

// ------------------------------ CAM gram -----------------------------------
// aa = x^T x per batch (512x512, K=4096); both operands from xT => pair loads.
__global__ void k_cam_gram(const unsigned short* __restrict__ xT,
                           float* __restrict__ aa) {
    int wave = (blockIdx.x * blockDim.x + threadIdx.x) >> 5;
    int lane = threadIdx.x & 31;
    int half = lane >> 4, ln16 = lane & 15;
    if (wave >= BATCH * 32 * 32) return;
    int b  = wave >> 10;
    int i0 = ((wave >> 5) & 31) << 4;
    int j0 = (wave & 31) << 4;

    const unsigned short* xbase = xT + (size_t)b * PLANE;   // [C][N]
    v8f acc = {};
    for (int k0 = 0; k0 < NTOK; k0 += 32) {
        v16bf a  = load_tile_pairs(xbase + (size_t)i0 * NTOK, NTOK, ln16, k0, half);
        v16bf bm = load_tile_pairs(xbase + (size_t)j0 * NTOK, NTOK, ln16, k0, half);
        acc = wmma_bf16(a, bm, acc);
    }
#pragma unroll
    for (int r = 0; r < 8; ++r) {
        int i = i0 + r + (half << 3);
        aa[(size_t)b * CCH * CCH + (size_t)i * CCH + j0 + ln16] = acc[r];
    }
}

// ------------------------ CAM row softmax (512 wide) -----------------------
// Writes the result transposed (softT[j][c]) so CAM apply gets pair loads.
__global__ void k_cam_softmax(const float* __restrict__ aa,
                              unsigned short* __restrict__ softT) {
    int wave = (blockIdx.x * blockDim.x + threadIdx.x) >> 5;
    int lane = threadIdx.x & 31;
    if (wave >= BATCH * CCH) return;
    int b = wave >> 9, c = wave & 511;
    const float* row = aa + (size_t)wave * CCH;

    float v[16];
    float m = -3.0e38f;
#pragma unroll
    for (int t = 0; t < 16; ++t) { v[t] = row[lane + t * 32]; m = fmaxf(m, v[t]); }
    m = fmaxf(m, __shfl_xor(m, 1));  m = fmaxf(m, __shfl_xor(m, 2));
    m = fmaxf(m, __shfl_xor(m, 4));  m = fmaxf(m, __shfl_xor(m, 8));
    m = fmaxf(m, __shfl_xor(m, 16));
    float s = 0.0f;
#pragma unroll
    for (int t = 0; t < 16; ++t) { v[t] = __expf(v[t] - m); s += v[t]; }
    s += __shfl_xor(s, 1); s += __shfl_xor(s, 2); s += __shfl_xor(s, 4);
    s += __shfl_xor(s, 8); s += __shfl_xor(s, 16);
    float inv = 1.0f / s;
    unsigned short* o = softT + (size_t)b * CCH * CCH + c;   // softT[j][c]
#pragma unroll
    for (int t = 0; t < 16; ++t)
        o[(size_t)(lane + t * 32) * CCH] = f2bf(v[t] * inv);
}

// ------------------- CAM apply + final fused epilogue ----------------------
// aaa = x @ soft; out += gamma_cam*aaa + 2*x.  B from softT => pair loads.
__global__ void k_cam_apply(const unsigned short* __restrict__ xb,
                            const unsigned short* __restrict__ softT,
                            const float* __restrict__ x,
                            const float* __restrict__ gamma_cam,
                            float* __restrict__ out) {
    int wave = (blockIdx.x * blockDim.x + threadIdx.x) >> 5;
    int lane = threadIdx.x & 31;
    int half = lane >> 4, ln16 = lane & 15;
    if (wave >= BATCH * (NTOK / 16) * 32) return;
    int ib = wave >> 5;
    int b  = ib >> 8;
    int n0 = (ib & 255) << 4;
    int j0 = (wave & 31) << 4;      // output channel tile

    v8f acc = {};
    for (int k0 = 0; k0 < CCH; k0 += 32) {
        v16bf a  = load_tile_pairs(xb + (size_t)(b * NTOK + n0) * CCH, CCH,
                                   ln16, k0, half);
        // B[k=c, n=j] = softT[j][c] : contiguous pairs along c
        v16bf bm = load_tile_pairs(softT + (size_t)b * CCH * CCH + (size_t)j0 * CCH,
                                   CCH, ln16, k0, half);
        acc = wmma_bf16(a, bm, acc);
    }
    float g = gamma_cam[0];
    int c = j0 + ln16;
#pragma unroll
    for (int r = 0; r < 8; ++r) {
        int n = n0 + r + (half << 3);
        size_t idx = (size_t)b * PLANE + (size_t)n * CCH + c;
        out[idx] += g * acc[r] + 2.0f * x[idx];   // (+x from PAM, +x from CAM)
    }
}

// ---------------------------------------------------------------------------
extern "C" void kernel_launch(void* const* d_in, const int* in_sizes, int n_in,
                              void* d_out, int out_size, void* d_ws, size_t ws_size,
                              hipStream_t stream) {
    (void)in_sizes; (void)n_in; (void)out_size; (void)ws_size;
    const float* x  = (const float*)d_in[0];
    const float* w1 = (const float*)d_in[1];
    const float* w2 = (const float*)d_in[2];
    const float* w3 = (const float*)d_in[3];
    const float* gp = (const float*)d_in[4];
    const float* gc = (const float*)d_in[5];
    float* out = (float*)d_out;

    char* ws = (char*)d_ws;
    size_t o = 0;
    unsigned short* xb    = (unsigned short*)(ws + o); o += (size_t)BATCH * PLANE * 2;   // 16 MB
    unsigned short* xT    = (unsigned short*)(ws + o); o += (size_t)BATCH * PLANE * 2;   // 16 MB
    unsigned short* w1T   = (unsigned short*)(ws + o); o += (size_t)CCH * K8 * 2;
    unsigned short* w2T   = (unsigned short*)(ws + o); o += (size_t)CCH * K8 * 2;
    unsigned short* w3T   = (unsigned short*)(ws + o); o += (size_t)CCH * CCH * 2;
    unsigned short* qm    = (unsigned short*)(ws + o); o += (size_t)BATCH * NTOK * K8 * 2;
    unsigned short* km    = (unsigned short*)(ws + o); o += (size_t)BATCH * NTOK * K8 * 2;
    unsigned short* dT    = (unsigned short*)(ws + o); o += (size_t)BATCH * NTOK * CCH * 2; // 16 MB
    float*          st    = (float*)(ws + o);          o += (size_t)BATCH * NTOK * 2 * 4;
    float*          aa    = (float*)(ws + o);          o += (size_t)BATCH * CCH * CCH * 4;
    unsigned short* softT = (unsigned short*)(ws + o); o += (size_t)BATCH * CCH * CCH * 2;

    const int T = 256;
    const int Mtot = BATCH * NTOK;          // 16384 token rows
    const int Mtiles = Mtot / 16;           // 1024

    // 1. bf16 conversions (+ transposed copies for pair-load layouts)
    k_convert_x <<<4096, T, 0, stream>>>(x, xb, xT);
    k_convert_wT<<<128,  T, 0, stream>>>(w1, w1T, CCH, K8);
    k_convert_wT<<<128,  T, 0, stream>>>(w2, w2T, CCH, K8);
    k_convert_wT<<<1024, T, 0, stream>>>(w3, w3T, CCH, CCH);

    // 2. projections: q,k row-major [M,64]; d transposed [C, M-batch-plane]
    k_gemm_bf16<<<(Mtiles * 4)  / 8, T, 0, stream>>>(xb, w1T, qm, Mtiles, 4,  CCH, K8, 0);
    k_gemm_bf16<<<(Mtiles * 4)  / 8, T, 0, stream>>>(xb, w2T, km, Mtiles, 4,  CCH, K8, 0);
    // dT: per-batch [C][N]; treating all batches as one [C, B*N] would interleave
    // batches, so launch per batch with M = NTOK.
    for (int b = 0; b < BATCH; ++b) {
        k_gemm_bf16<<<((NTOK / 16) * 32) / 8, T, 0, stream>>>(
            xb + (size_t)b * PLANE, w3T, dT + (size_t)b * PLANE,
            NTOK / 16, 32, CCH, NTOK, 1);
    }

    // 3. PAM softmax row stats (flash-style pass 1)
    k_pam_stats<<<(BATCH * NTOK / 16) / 8, T, 0, stream>>>(qm, km, st);

    // 4. PAM attention apply + scrambled reshape epilogue (writes every out elem)
    k_pam_apply<<<BATCH * NTOK / 16, T, 0, stream>>>(qm, km, dT, st, gp, out);

    // 5-7. CAM: gram, softmax(->transposed), apply (+ final fused accumulation)
    k_cam_gram   <<<(BATCH * 32 * 32) / 8, T, 0, stream>>>(xT, aa);
    k_cam_softmax<<<(BATCH * CCH) / 8,     T, 0, stream>>>(aa, softT);
    k_cam_apply  <<<(BATCH * (NTOK / 16) * 32) / 8, T, 0, stream>>>(xb, softT, x, gc, out);
}
